// MOE_49658411876490
// MI455X (gfx1250) — matrix operand order repaired and checked by reading, not compiled, not run on hardware
//
#include <hip/hip_runtime.h>
#include <stdint.h>
#include <math.h>

// ---------------- problem constants ----------------
#define B_    4
#define S_    2048
#define D_    1024
#define H_    4096
#define E_    8
#define CAP_  2048
#define N_    (B_ * S_)      // 8192 tokens
#define ALPHA_ 0.01f

// ---------------- WMMA / TDM types (gfx1250, wave32) ----------------
typedef __attribute__((ext_vector_type(16))) __bf16 v16bf;
typedef __attribute__((ext_vector_type(8)))  float  v8f;
typedef __attribute__((ext_vector_type(4)))  unsigned int v4u;
typedef __attribute__((ext_vector_type(8)))  unsigned int v8u;

__device__ __forceinline__ unsigned short f2bf(float f) {
    union { float f; unsigned int u; } c; c.f = f;
    unsigned int u = c.u;
    unsigned int r = (u + 0x7FFFu + ((u >> 16) & 1u)) >> 16;   // RNE
    return (unsigned short)r;
}

// ---------------- GEMM tiling ----------------
#define BM 128
#define BN 128
#define BK 32
#define LDT 40   // padded K-stride (bf16 elems) == TDM row stride 64B + 16B pad

// =====================================================================
// TDM: stage a 2D bf16 tile (tile_k=32 contiguous x tile_n=128 rows) from a
// row-major [Ndim][Kdim] bf16 tensor into LDS with 80B padded row stride.
// D# per CDNA5 ISA. Issued per-wave (EXEC ignored).
// =====================================================================
__device__ __forceinline__ void tdm_tile_load(unsigned lds_off,
                                              const unsigned short* gptr,
                                              unsigned Kdim, unsigned Ndim)
{
    unsigned long long ga = (unsigned long long)(uintptr_t)gptr;
    v4u g0;
    g0[0] = 0x1u;                                    // count=1, user mode
    g0[1] = lds_off;                                 // lds_addr (bytes)
    g0[2] = (unsigned)(ga & 0xffffffffu);            // global_addr[31:0]
    g0[3] = (unsigned)((ga >> 32) & 0x01ffffffu)     // global_addr[56:32]
          | (2u << 30);                              // type = 2 ("image")
    v8u g1;
    g1[0] = (1u << 16)                               // data_size = 2 bytes
          | (1u << 20)                               // pad_enable
          | (3u << 22)                               // pad_interval: 16 DWORDs (64B)
          | (3u << 25);                              // pad_amount:   4 DWORDs (16B)
    g1[1] = (Kdim & 0xffffu) << 16;                  // tensor_dim0[15:0]
    g1[2] = ((Kdim >> 16) & 0xffffu)                 // tensor_dim0[31:16]
          | ((Ndim & 0xffffu) << 16);                // tensor_dim1[15:0]
    g1[3] = ((Ndim >> 16) & 0xffffu)                 // tensor_dim1[31:16]
          | (32u << 16);                             // tile_dim0 = 32 elems (64B)
    g1[4] = 128u;                                    // tile_dim1 = 128 rows (tile_dim2=0)
    g1[5] = Kdim;                                    // tensor_dim0_stride[31:0]
    g1[6] = 0u;                                      // stride hi / dim1_stride lo
    g1[7] = 0u;
    asm volatile("tensor_load_to_lds %0, %1" :: "s"(g0), "s"(g1) : "memory");
}

// =====================================================================
// Prep 0: zero output accumulator
// =====================================================================
__global__ void zero_kernel(float* __restrict__ p, long long n) {
    long long i = (long long)blockIdx.x * blockDim.x + threadIdx.x;
    if (i < n) p[i] = 0.0f;
}

// =====================================================================
// Prep 1: elementwise fp32 -> bf16 (activations)
// =====================================================================
__global__ __launch_bounds__(256)
void cvt_bf16_kernel(const float* __restrict__ src,
                     unsigned short* __restrict__ dst, long long n) {
    long long i = ((long long)blockIdx.x * blockDim.x + threadIdx.x) * 4;
    if (i + 3 < n) {
        float4 v = *(const float4*)(src + i);
        ushort4 o;
        o.x = f2bf(v.x); o.y = f2bf(v.y); o.z = f2bf(v.z); o.w = f2bf(v.w);
        *(ushort4*)(dst + i) = o;
    }
}

// =====================================================================
// Prep 2: transpose-convert  src fp32 [K][Nd]  ->  dst bf16 [Nd][K]
// (batched over blockIdx.z; 64x64 tiles through LDS)
// =====================================================================
__global__ __launch_bounds__(256)
void transpose_cvt_kernel(const float* __restrict__ src,
                          unsigned short* __restrict__ dst, int K, int Nd) {
    __shared__ __align__(16) unsigned short T[64][72];
    const long long bo = (long long)blockIdx.z * K * Nd;
    const int k0 = blockIdx.y * 64;
    const int n0 = blockIdx.x * 64;
    const int tid = threadIdx.x;

    const int kl = tid >> 2;
    const int nb = (tid & 3) * 16;
    const float* srow = src + bo + (long long)(k0 + kl) * Nd + n0 + nb;
    for (int j = 0; j < 16; j += 4) {
        float4 v = *(const float4*)(srow + j);
        T[nb + j + 0][kl] = f2bf(v.x);
        T[nb + j + 1][kl] = f2bf(v.y);
        T[nb + j + 2][kl] = f2bf(v.z);
        T[nb + j + 3][kl] = f2bf(v.w);
    }
    __syncthreads();

    const int nl = tid >> 2;
    const int kb = (tid & 3) * 16;
    unsigned short* drow = dst + bo + (long long)(n0 + nl) * K + k0 + kb;
    *(uint4*)(drow + 0) = *(const uint4*)&T[nl][kb + 0];
    *(uint4*)(drow + 8) = *(const uint4*)&T[nl][kb + 8];
}

// =====================================================================
// Router: softmax(x@rw+rb), top-2, renormalize. One wave per token.
// =====================================================================
__global__ __launch_bounds__(256)
void router_kernel(const float* __restrict__ x,
                   const float* __restrict__ rw,
                   const float* __restrict__ rb,
                   int*   __restrict__ ti,
                   float* __restrict__ tw,
                   float* __restrict__ w_of)
{
    __shared__ float rws[D_ * E_];   // 32 KB
    const int tid  = threadIdx.x;
    const int lane = tid & 31;
    const int wid  = tid >> 5;

    for (int i = tid; i < D_ * E_; i += 256) rws[i] = rw[i];
    __syncthreads();

    const int n = blockIdx.x * 8 + wid;
    if (n >= N_) return;

    float acc[E_];
#pragma unroll
    for (int e = 0; e < E_; ++e) acc[e] = 0.0f;
    const float* xp = x + (long long)n * D_;
    for (int d = lane; d < D_; d += 32) {
        float xv = xp[d];
        const float* r = &rws[d * E_];
#pragma unroll
        for (int e = 0; e < E_; ++e) acc[e] += xv * r[e];
    }
#pragma unroll
    for (int off = 16; off > 0; off >>= 1)
#pragma unroll
        for (int e = 0; e < E_; ++e) acc[e] += __shfl_xor(acc[e], off, 32);

    if (lane == 0) {
        float m = -1e30f;
#pragma unroll
        for (int e = 0; e < E_; ++e) { acc[e] += rb[e]; m = fmaxf(m, acc[e]); }
        float p[E_]; float s = 0.0f;
#pragma unroll
        for (int e = 0; e < E_; ++e) { p[e] = __expf(acc[e] - m); s += p[e]; }
        int i0 = 0;
#pragma unroll
        for (int e = 1; e < E_; ++e) if (p[e] > p[i0]) i0 = e;
        int i1 = (i0 == 0) ? 1 : 0;
#pragma unroll
        for (int e = 0; e < E_; ++e) if (e != i0 && p[e] > p[i1]) i1 = e;
        float t0 = p[i0], t1 = p[i1];
        float inv = 1.0f / (t0 + t1);
        ti[n * 2 + 0] = i0;  tw[n * 2 + 0] = t0 * inv;
        ti[n * 2 + 1] = i1;  tw[n * 2 + 1] = t1 * inv;
        w_of[n] = 0.0f;
    }
}

// =====================================================================
// Deterministic per-expert ordered scan -> dispatch buffers.
// =====================================================================
__global__ __launch_bounds__(1024)
void dispatch_scan_kernel(const int*   __restrict__ ti,
                          const float* __restrict__ tw,
                          int*   __restrict__ buf,
                          float* __restrict__ wc,
                          float* __restrict__ w_of,
                          int*   __restrict__ cnt,
                          float* __restrict__ sumw)
{
    const int e   = blockIdx.x;
    const int tid = threadIdx.x;
    __shared__ int   sc[1024];
    __shared__ int   carry_s;
    __shared__ float sumw_s;
    if (tid == 0) { carry_s = 0; sumw_s = 0.0f; }
    __syncthreads();

    for (int t = 0; t < N_ / 1024; ++t) {
        const int n  = t * 1024 + tid;
        const int i0 = ti[n * 2 + 0];
        const int i1 = ti[n * 2 + 1];
        const float w = (i0 == e) ? tw[n * 2 + 0] : ((i1 == e) ? tw[n * 2 + 1] : 0.0f);
        const int mi  = (i0 == e || i1 == e) ? 1 : 0;

        sc[tid] = mi;
        __syncthreads();
        for (int off = 1; off < 1024; off <<= 1) {
            int v = (tid >= off) ? sc[tid - off] : 0;
            __syncthreads();
            sc[tid] += v;
            __syncthreads();
        }
        const int incl = sc[tid];
        const int rank = carry_s + incl - mi;
        if (mi) {
            if (rank < CAP_) {
                buf[e * CAP_ + rank] = n;
                wc [e * CAP_ + rank] = w;
                atomicAdd(&sumw_s, w);
            } else {
                atomicAdd(&w_of[n], w);
            }
        }
        __syncthreads();
        if (tid == 1023) carry_s += incl;
        __syncthreads();
    }

    const int total = carry_s;
    for (int s = total + tid; s < CAP_; s += 1024) {
        buf[e * CAP_ + s] = N_;
        wc [e * CAP_ + s] = 0.0f;
    }
    if (tid == 0) {
        cnt [e] = (total < CAP_) ? total : CAP_;
        sumw[e] = sumw_s;
    }
}

// =====================================================================
// Aux loss scalar
// =====================================================================
__global__ void aux_kernel(const int* __restrict__ cnt,
                           const float* __restrict__ sumw,
                           float* __restrict__ aux_out)
{
    if (threadIdx.x == 0 && blockIdx.x == 0) {
        float a = 0.0f;
#pragma unroll
        for (int e = 0; e < E_; ++e) {
            float f   = (float)cnt[e] / (float)N_;
            float den = (float)(cnt[e] > 1 ? cnt[e] : 1);
            a += f * (sumw[e] / den);
        }
        *aux_out = ALPHA_ * (float)E_ * a;
    }
}

// =====================================================================
// WMMA fragment loads from LDS tiles stored [row][K] with stride LDT.
// Contiguous k order, same for A and B: lane L, kh=L>>4, lm=L&15.
//   A frag: M=lm, K run 16kh..16kh+15
//   B frag: N=lm, K run 16kh..16kh+15
// =====================================================================
__device__ __forceinline__ v16bf load_a_frag(const unsigned short* T, int mbase,
                                             int lm, int kh) {
    union { v16bf v; uint4 q[2]; } u;
    const unsigned short* ap = T + (mbase + lm) * LDT;
    u.q[0] = *(const uint4*)(ap + 16 * kh);
    u.q[1] = *(const uint4*)(ap + 16 * kh + 8);
    return u.v;
}
__device__ __forceinline__ v16bf load_b_frag(const unsigned short* T, int nbase,
                                             int lm, int kh) {
    union { v16bf v; uint4 q[2]; } u;
    const unsigned short* bp = T + (nbase + lm) * LDT + 16 * kh;
    u.q[0] = *(const uint4*)(bp + 0);
    u.q[1] = *(const uint4*)(bp + 8);
    return u.v;
}

#define WMMA_BF16(accv, av, bv) \
    accv = __builtin_amdgcn_wmma_f32_16x16x32_bf16(false, av, false, bv, (short)0, accv, false, false)

// =====================================================================
// Expert FFN1: h[e,c,:] = relu(xb[buf[e,c]] @ w1t[e]^T + b1[e])
// A: bf16 token-row gather (manual). B: TDM-staged from w1t (E,H,D).
// =====================================================================
__global__ __launch_bounds__(128)
void expert_ffn1_kernel(const unsigned short* __restrict__ xb,   // (N,D) bf16
                        const unsigned short* __restrict__ w1t,  // (E,H,D) bf16
                        const float* __restrict__ b1,            // (E,H)
                        const int*   __restrict__ buf,
                        unsigned short* __restrict__ h)          // (E,CAP,H) bf16
{
    const int e  = blockIdx.z;
    const int c0 = blockIdx.y * BM;
    const int n0 = blockIdx.x * BN;
    const int tid  = threadIdx.x;
    const int lane = tid & 31;
    const int wid  = tid >> 5;
    const int wrow = (wid >> 1) * 64;
    const int wcol = (wid & 1) * 64;
    const int lm = lane & 15, kh = lane >> 4;

    __shared__ __align__(16) unsigned short As[BM * LDT];
    __shared__ __align__(16) unsigned short Bs[BN * LDT];
    __shared__ int rowtok[BM];

    for (int i = tid; i < BM; i += 128) rowtok[i] = buf[e * CAP_ + c0 + i];
    __syncthreads();

    const unsigned bs_off = (unsigned)(uintptr_t)(void*)Bs;
    const unsigned short* w1te = w1t + (long long)e * H_ * D_;

    v8f acc[4][4];
#pragma unroll
    for (int mi = 0; mi < 4; ++mi)
#pragma unroll
        for (int ni = 0; ni < 4; ++ni) acc[mi][ni] = (v8f)(0.0f);

    for (int kk = 0; kk < D_; kk += BK) {
        if (tid < 32)   // wave 0 drives the Tensor Data Mover for B
            tdm_tile_load(bs_off, w1te + (long long)n0 * D_ + kk, D_, H_);

        // A: gathered bf16 token rows, one row (64B) per thread
        {
            const int tok = rowtok[tid];
            uint4 v0, v1, v2, v3;
            if (tok < N_) {
                const unsigned short* s = xb + (long long)tok * D_ + kk;
                v0 = *(const uint4*)(s + 0);  v1 = *(const uint4*)(s + 8);
                v2 = *(const uint4*)(s + 16); v3 = *(const uint4*)(s + 24);
            } else {
                v0 = v1 = v2 = v3 = make_uint4(0u, 0u, 0u, 0u);
            }
            unsigned short* d = &As[tid * LDT];
            *(uint4*)(d + 0)  = v0; *(uint4*)(d + 8)  = v1;
            *(uint4*)(d + 16) = v2; *(uint4*)(d + 24) = v3;
        }

        if (tid < 32) __builtin_amdgcn_s_wait_tensorcnt(0);
        __syncthreads();

        v16bf a[4];
#pragma unroll
        for (int mi = 0; mi < 4; ++mi) a[mi] = load_a_frag(As, wrow + mi * 16, lm, kh);
#pragma unroll
        for (int ni = 0; ni < 4; ++ni) {
            v16bf b = load_b_frag(Bs, wcol + ni * 16, lm, kh);
#pragma unroll
            for (int mi = 0; mi < 4; ++mi) WMMA_BF16(acc[mi][ni], a[mi], b);
        }
        __syncthreads();
    }

#pragma unroll
    for (int ni = 0; ni < 4; ++ni) {
        const int n = n0 + wcol + ni * 16 + lm;
        const float bias = b1[e * H_ + n];
#pragma unroll
        for (int mi = 0; mi < 4; ++mi) {
            const int mb = c0 + wrow + mi * 16 + 8 * kh;
#pragma unroll
            for (int r = 0; r < 8; ++r) {
                float v = acc[mi][ni][r] + bias;
                v = v > 0.0f ? v : 0.0f;
                h[((long long)e * CAP_ + mb + r) * H_ + n] = f2bf(v);
            }
        }
    }
}

// =====================================================================
// Expert FFN2 + combine: out[tok] += wc * (h[e] @ w2t[e]^T + b2[e])
// A and B both TDM-staged.
// =====================================================================
__global__ __launch_bounds__(128)
void expert_ffn2_kernel(const unsigned short* __restrict__ h,    // (E,CAP,H) bf16
                        const unsigned short* __restrict__ w2t,  // (E,D,H) bf16
                        const float* __restrict__ b2,            // (E,D)
                        const int*   __restrict__ buf,
                        const float* __restrict__ wc,
                        float* __restrict__ out)                 // (N,D)
{
    const int e  = blockIdx.z;
    const int c0 = blockIdx.y * BM;
    const int n0 = blockIdx.x * BN;
    const int tid  = threadIdx.x;
    const int lane = tid & 31;
    const int wid  = tid >> 5;
    const int wrow = (wid >> 1) * 64;
    const int wcol = (wid & 1) * 64;
    const int lm = lane & 15, kh = lane >> 4;

    __shared__ __align__(16) unsigned short As[BM * LDT];
    __shared__ __align__(16) unsigned short Bs[BN * LDT];
    __shared__ int   rowtok[BM];
    __shared__ float roww[BM];

    for (int i = tid; i < BM; i += 128) {
        rowtok[i] = buf[e * CAP_ + c0 + i];
        roww[i]   = wc [e * CAP_ + c0 + i];
    }
    __syncthreads();

    const unsigned as_off = (unsigned)(uintptr_t)(void*)As;
    const unsigned bs_off = (unsigned)(uintptr_t)(void*)Bs;
    const unsigned short* he   = h   + (long long)e * CAP_ * H_;
    const unsigned short* w2te = w2t + (long long)e * D_ * H_;

    v8f acc[4][4];
#pragma unroll
    for (int mi = 0; mi < 4; ++mi)
#pragma unroll
        for (int ni = 0; ni < 4; ++ni) acc[mi][ni] = (v8f)(0.0f);

    for (int kk = 0; kk < H_; kk += BK) {
        if (tid < 32) {
            tdm_tile_load(as_off, he   + (long long)c0 * H_ + kk, H_, CAP_);
            tdm_tile_load(bs_off, w2te + (long long)n0 * H_ + kk, H_, D_);
            __builtin_amdgcn_s_wait_tensorcnt(0);
        }
        __syncthreads();

        v16bf a[4];
#pragma unroll
        for (int mi = 0; mi < 4; ++mi) a[mi] = load_a_frag(As, wrow + mi * 16, lm, kh);
#pragma unroll
        for (int ni = 0; ni < 4; ++ni) {
            v16bf b = load_b_frag(Bs, wcol + ni * 16, lm, kh);
#pragma unroll
            for (int mi = 0; mi < 4; ++mi) WMMA_BF16(acc[mi][ni], a[mi], b);
        }
        __syncthreads();
    }

#pragma unroll
    for (int ni = 0; ni < 4; ++ni) {
        const int n = n0 + wcol + ni * 16 + lm;
        const float bias = b2[e * D_ + n];
#pragma unroll
        for (int mi = 0; mi < 4; ++mi) {
            const int lrow = wrow + mi * 16 + 8 * kh;
#pragma unroll
            for (int r = 0; r < 8; ++r) {
                const int tok = rowtok[lrow + r];
                if (tok < N_) {
                    const float v = roww[lrow + r] * (acc[mi][ni][r] + bias);
                    atomicAdd(&out[(long long)tok * D_ + n], v);
                }
            }
        }
    }
}

// =====================================================================
// Shared MLP layer 1: hs = relu(xb @ ow1t^T + ob1).  A and B TDM-staged.
// =====================================================================
__global__ __launch_bounds__(128)
void shared_ffn1_kernel(const unsigned short* __restrict__ xb,    // (N,D) bf16
                        const unsigned short* __restrict__ ow1t,  // (H,D) bf16
                        const float* __restrict__ ob1,
                        unsigned short* __restrict__ hs)          // (N,H) bf16
{
    const int r0 = blockIdx.y * BM;
    const int n0 = blockIdx.x * BN;
    const int tid  = threadIdx.x;
    const int lane = tid & 31;
    const int wid  = tid >> 5;
    const int wrow = (wid >> 1) * 64;
    const int wcol = (wid & 1) * 64;
    const int lm = lane & 15, kh = lane >> 4;

    __shared__ __align__(16) unsigned short As[BM * LDT];
    __shared__ __align__(16) unsigned short Bs[BN * LDT];
    const unsigned as_off = (unsigned)(uintptr_t)(void*)As;
    const unsigned bs_off = (unsigned)(uintptr_t)(void*)Bs;

    v8f acc[4][4];
#pragma unroll
    for (int mi = 0; mi < 4; ++mi)
#pragma unroll
        for (int ni = 0; ni < 4; ++ni) acc[mi][ni] = (v8f)(0.0f);

    for (int kk = 0; kk < D_; kk += BK) {
        if (tid < 32) {
            tdm_tile_load(as_off, xb   + (long long)r0 * D_ + kk, D_, N_);
            tdm_tile_load(bs_off, ow1t + (long long)n0 * D_ + kk, D_, H_);
            __builtin_amdgcn_s_wait_tensorcnt(0);
        }
        __syncthreads();

        v16bf a[4];
#pragma unroll
        for (int mi = 0; mi < 4; ++mi) a[mi] = load_a_frag(As, wrow + mi * 16, lm, kh);
#pragma unroll
        for (int ni = 0; ni < 4; ++ni) {
            v16bf b = load_b_frag(Bs, wcol + ni * 16, lm, kh);
#pragma unroll
            for (int mi = 0; mi < 4; ++mi) WMMA_BF16(acc[mi][ni], a[mi], b);
        }
        __syncthreads();
    }

#pragma unroll
    for (int ni = 0; ni < 4; ++ni) {
        const int n = n0 + wcol + ni * 16 + lm;
        const float bias = ob1[n];
#pragma unroll
        for (int mi = 0; mi < 4; ++mi) {
            const int mb = r0 + wrow + mi * 16 + 8 * kh;
#pragma unroll
            for (int r = 0; r < 8; ++r) {
                float v = acc[mi][ni][r] + bias;
                v = v > 0.0f ? v : 0.0f;
                hs[(long long)(mb + r) * H_ + n] = f2bf(v);
            }
        }
    }
}

// =====================================================================
// Shared MLP layer 2: out[n] += w_of[n] * (hs @ ow2t^T + ob2)
// Runs after expert combine; unique writer per element -> plain RMW.
// =====================================================================
__global__ __launch_bounds__(128)
void shared_ffn2_kernel(const unsigned short* __restrict__ hs,    // (N,H) bf16
                        const unsigned short* __restrict__ ow2t,  // (D,H) bf16
                        const float* __restrict__ ob2,
                        const float* __restrict__ w_of,
                        float* __restrict__ out)
{
    const int r0 = blockIdx.y * BM;
    const int n0 = blockIdx.x * BN;
    const int tid  = threadIdx.x;
    const int lane = tid & 31;
    const int wid  = tid >> 5;
    const int wrow = (wid >> 1) * 64;
    const int wcol = (wid & 1) * 64;
    const int lm = lane & 15, kh = lane >> 4;

    __shared__ __align__(16) unsigned short As[BM * LDT];
    __shared__ __align__(16) unsigned short Bs[BN * LDT];
    const unsigned as_off = (unsigned)(uintptr_t)(void*)As;
    const unsigned bs_off = (unsigned)(uintptr_t)(void*)Bs;

    v8f acc[4][4];
#pragma unroll
    for (int mi = 0; mi < 4; ++mi)
#pragma unroll
        for (int ni = 0; ni < 4; ++ni) acc[mi][ni] = (v8f)(0.0f);

    for (int kk = 0; kk < H_; kk += BK) {
        if (tid < 32) {
            tdm_tile_load(as_off, hs   + (long long)r0 * H_ + kk, H_, N_);
            tdm_tile_load(bs_off, ow2t + (long long)n0 * H_ + kk, H_, D_);
            __builtin_amdgcn_s_wait_tensorcnt(0);
        }
        __syncthreads();

        v16bf a[4];
#pragma unroll
        for (int mi = 0; mi < 4; ++mi) a[mi] = load_a_frag(As, wrow + mi * 16, lm, kh);
#pragma unroll
        for (int ni = 0; ni < 4; ++ni) {
            v16bf b = load_b_frag(Bs, wcol + ni * 16, lm, kh);
#pragma unroll
            for (int mi = 0; mi < 4; ++mi) WMMA_BF16(acc[mi][ni], a[mi], b);
        }
        __syncthreads();
    }

#pragma unroll
    for (int ni = 0; ni < 4; ++ni) {
        const int n = n0 + wcol + ni * 16 + lm;
        const float bias = ob2[n];
#pragma unroll
        for (int mi = 0; mi < 4; ++mi) {
            const int mb = r0 + wrow + mi * 16 + 8 * kh;
#pragma unroll
            for (int r = 0; r < 8; ++r) {
                const int row = mb + r;
                float* op = &out[(long long)row * D_ + n];
                *op += w_of[row] * (acc[mi][ni][r] + bias);
            }
        }
    }
}

// =====================================================================
// Host launcher
// =====================================================================
extern "C" void kernel_launch(void* const* d_in, const int* in_sizes, int n_in,
                              void* d_out, int out_size, void* d_ws, size_t ws_size,
                              hipStream_t stream)
{
    (void)in_sizes; (void)n_in; (void)out_size; (void)ws_size;

    const float* x   = (const float*)d_in[0];
    const float* rw  = (const float*)d_in[1];
    const float* rb  = (const float*)d_in[2];
    const float* w1  = (const float*)d_in[3];
    const float* b1  = (const float*)d_in[4];
    const float* w2  = (const float*)d_in[5];
    const float* b2  = (const float*)d_in[6];
    const float* ow1 = (const float*)d_in[7];
    const float* ob1 = (const float*)d_in[8];
    const float* ow2 = (const float*)d_in[9];
    const float* ob2 = (const float*)d_in[10];

    float* out = (float*)d_out;
    float* aux = out + (long long)N_ * D_;

    // ---- workspace carve-up ----
    char* ws = (char*)d_ws;
    size_t off = 0;
    int*   ti   = (int*)  (ws + off); off += (size_t)N_ * 2 * sizeof(int);
    float* tw   = (float*)(ws + off); off += (size_t)N_ * 2 * sizeof(float);
    int*   buf  = (int*)  (ws + off); off += (size_t)E_ * CAP_ * sizeof(int);
    float* wc   = (float*)(ws + off); off += (size_t)E_ * CAP_ * sizeof(float);
    float* w_of = (float*)(ws + off); off += (size_t)N_ * sizeof(float);
    int*   cnt  = (int*)  (ws + off); off += (size_t)E_ * sizeof(int);
    float* sumw = (float*)(ws + off); off += (size_t)E_ * sizeof(float);
    off = (off + 255) & ~(size_t)255;
    unsigned short* xb   = (unsigned short*)(ws + off); off += (size_t)N_ * D_ * 2;       // 16 MB
    unsigned short* w1t  = (unsigned short*)(ws + off); off += (size_t)E_ * H_ * D_ * 2;  // 64 MB
    unsigned short* w2t  = (unsigned short*)(ws + off); off += (size_t)E_ * D_ * H_ * 2;  // 64 MB
    unsigned short* ow1t = (unsigned short*)(ws + off); off += (size_t)H_ * D_ * 2;       //  8 MB
    unsigned short* ow2t = (unsigned short*)(ws + off); off += (size_t)D_ * H_ * 2;       //  8 MB
    unsigned short* hexp = (unsigned short*)(ws + off); off += (size_t)E_ * CAP_ * H_ * 2;// 128 MB
    unsigned short* hsh  = (unsigned short*)(ws + off); off += (size_t)N_ * H_ * 2;       //  64 MB

    // 0) zero output accumulator
    {
        long long tot = (long long)N_ * D_;
        zero_kernel<<<dim3((unsigned)((tot + 255) / 256)), dim3(256), 0, stream>>>(out, tot);
    }
    // 1) bf16 prep: activations + transposed weights
    cvt_bf16_kernel<<<dim3((unsigned)((long long)N_ * D_ / 1024)), dim3(256), 0, stream>>>(
        x, xb, (long long)N_ * D_);
    transpose_cvt_kernel<<<dim3(H_ / 64, D_ / 64, E_), dim3(256), 0, stream>>>(w1, w1t, D_, H_);
    transpose_cvt_kernel<<<dim3(D_ / 64, H_ / 64, E_), dim3(256), 0, stream>>>(w2, w2t, H_, D_);
    transpose_cvt_kernel<<<dim3(H_ / 64, D_ / 64, 1),  dim3(256), 0, stream>>>(ow1, ow1t, D_, H_);
    transpose_cvt_kernel<<<dim3(D_ / 64, H_ / 64, 1),  dim3(256), 0, stream>>>(ow2, ow2t, H_, D_);
    // 2) router (also zeros w_of)
    router_kernel<<<dim3(N_ / 8), dim3(256), 0, stream>>>(x, rw, rb, ti, tw, w_of);
    // 3) deterministic dispatch scan
    dispatch_scan_kernel<<<dim3(E_), dim3(1024), 0, stream>>>(ti, tw, buf, wc, w_of, cnt, sumw);
    // 4) aux scalar
    aux_kernel<<<dim3(1), dim3(32), 0, stream>>>(cnt, sumw, aux);
    // 5) expert FFN1
    expert_ffn1_kernel<<<dim3(H_ / BN, CAP_ / BM, E_), dim3(128), 0, stream>>>(
        xb, w1t, b1, buf, hexp);
    // 6) expert FFN2 + combine
    expert_ffn2_kernel<<<dim3(D_ / BN, CAP_ / BM, E_), dim3(128), 0, stream>>>(
        hexp, w2t, b2, buf, wc, out);
    // 7) shared FFN1
    shared_ffn1_kernel<<<dim3(H_ / BN, N_ / BM), dim3(128), 0, stream>>>(xb, ow1t, ob1, hsh);
    // 8) shared FFN2 + overflow add
    shared_ffn2_kernel<<<dim3(D_ / BN, N_ / BM), dim3(128), 0, stream>>>(hsh, ow2t, ob2, w_of, out);
}
